// TemporalSSIMLoss_60103772340291
// MI455X (gfx1250) — matrix operand order repaired and checked by reading, not compiled, not run on hardware
//
#include <hip/hip_runtime.h>

// ---------------------------------------------------------------------------
// TemporalSSIMLoss on MI455X (gfx1250, wave32, WMMA)
//
// 11-tap Gaussian conv over a 16-wide tile == one v_wmma_f32_16x16x32_f16
// with a Toeplitz B matrix (K=26 <= 32). Five convs (p, q, p^2, q^2, pq)
// -> 5 WMMAs per 16x16 tile, f32 accumulators. Inputs staged global->LDS
// as f16 with nontemporal loads (each HBM byte read once; halo from LDS).
// Toeplitz B built via a 1KB LDS table (no cndmask chains). SSIM pointwise
// in f32 with v_rcp_f32. Deterministic two-stage reduction (no atomics).
// ---------------------------------------------------------------------------

typedef _Float16 v16h __attribute__((ext_vector_type(16)));
typedef _Float16 v8h  __attribute__((ext_vector_type(8)));
typedef float    v8f  __attribute__((ext_vector_type(8)));

#define T_LEN    8192
#define N_ROWS   2048              // B*C = 32*64
#define WIN      11
#define HALO     5
#define T_CHUNK  512               // output cols per workgroup
#define COLS     (T_CHUNK + 16)    // 528 staged cols incl. halo (1056B rows, 16B aligned)
#define ROWS16   16
#define THREADS  256               // 8 waves of 32
#define TILES_PER_WAVE 4           // 32 tiles / 8 waves
#define SSIM_C1  4.0e-4f           // (0.01*2)^2
#define SSIM_C2  3.6e-3f           // (0.03*2)^2
#define TOTAL_INV (1.0f / 16777216.0f)   // 1 / (2048*8192)

// Normalized Gaussian window, sigma = 11/6 (compile-time; f16-B anyway).
__device__ __constant__ const float G_WIN[WIN] = {
    0.0052911f, 0.0201843f, 0.0571815f, 0.1203049f, 0.1879752f,
    0.2181254f,
    0.1879752f, 0.1203049f, 0.0571815f, 0.0201843f, 0.0052911f
};

__global__ __launch_bounds__(THREADS)
void ssim_wmma_kernel(const float* __restrict__ pred,
                      const float* __restrict__ target,
                      float* __restrict__ ws)
{
    __shared__ __align__(16) _Float16 pT[ROWS16][COLS];
    __shared__ __align__(16) _Float16 qT[ROWS16][COLS];
    __shared__ __align__(16) _Float16 Bt[16][32];   // Toeplitz, Bt[n][k] = w[k-n]
    __shared__ float red[THREADS];

    const int tid         = threadIdx.x;
    const int chunk_start = blockIdx.x * T_CHUNK;
    const int row0        = blockIdx.y * ROWS16;

    // ---- Fill Toeplitz table (512 halves, 2 per thread)
    #pragma unroll
    for (int idx = tid; idx < 16 * 32; idx += THREADS) {
        int n = idx >> 5, k = idx & 31;
        int d = k - n;
        Bt[n][k] = ((unsigned)d < WIN) ? (_Float16)G_WIN[d] : (_Float16)0.f;
    }

    // ---- Stage 16 x 528 of p and q into LDS as f16
    const bool edge = (blockIdx.x == 0) || (blockIdx.x == gridDim.x - 1);
    if (!edge) {
        for (int idx = tid; idx < ROWS16 * COLS; idx += THREADS) {
            int r = idx / COLS;
            int u = idx - r * COLS;
            size_t off = (size_t)(row0 + r) * T_LEN + (size_t)(chunk_start + u - HALO);
            pT[r][u] = (_Float16)__builtin_nontemporal_load(pred + off);
            qT[r][u] = (_Float16)__builtin_nontemporal_load(target + off);
        }
    } else {
        for (int idx = tid; idx < ROWS16 * COLS; idx += THREADS) {
            int r = idx / COLS;
            int u = idx - r * COLS;
            int g = chunk_start + u - HALO;                 // reflect at global edges
            if (g < 0)            g = -g;
            else if (g >= T_LEN)  g = 2 * (T_LEN - 1) - g;
            size_t off = (size_t)(row0 + r) * T_LEN + (size_t)g;
            pT[r][u] = (_Float16)__builtin_nontemporal_load(pred + off);
            qT[r][u] = (_Float16)__builtin_nontemporal_load(target + off);
        }
    }
    __syncthreads();

    const int lane = tid & 31;
    const int wave = tid >> 5;
    const int m    = lane & 15;                 // A-matrix row / B column for this lane
    const int hb   = (lane < 16) ? 0 : 8;       // A half-K base selector
    const int kb   = (lane < 16) ? 0 : 16;      // B K base

    // ---- B fragment from LDS table: lane<16: K {0..7,16..23}; lane>=16: K {8..15,24..31} (matches A)
    v16h bw;
    {
        v8h b_lo = *(const v8h*)&Bt[m][hb];
        v8h b_hi = *(const v8h*)&Bt[m][hb + 16];
        #pragma unroll
        for (int i = 0; i < 8; ++i) { bw[i] = b_lo[i]; bw[i + 8] = b_hi[i]; }
    }

    float acc = 0.f;
    const v8f zero = {};

    #pragma unroll 2
    for (int tt = 0; tt < TILES_PER_WAVE; ++tt) {
        const int tile = wave * TILES_PER_WAVE + tt;
        // 16-bit A 16x32 layout: lane<16 holds K {0..7,16..23},
        // lane>=16 holds K {8..15,24..31} -> contiguous 16B runs in time dim.
        const int ub = tile * 16 + hb;

        v8h p_lo = *(const v8h*)&pT[m][ub];
        v8h p_hi = *(const v8h*)&pT[m][ub + 16];
        v8h q_lo = *(const v8h*)&qT[m][ub];
        v8h q_hi = *(const v8h*)&qT[m][ub + 16];

        v16h ap, aq;
        #pragma unroll
        for (int i = 0; i < 8; ++i) {
            ap[i] = p_lo[i]; ap[i + 8] = p_hi[i];
            aq[i] = q_lo[i]; aq[i + 8] = q_hi[i];
        }

        v8f mu_x = __builtin_amdgcn_wmma_f32_16x16x32_f16(false, ap, false, bw,
                                                          (short)0, zero, false, false);
        v8f mu_y = __builtin_amdgcn_wmma_f32_16x16x32_f16(false, aq, false, bw,
                                                          (short)0, zero, false, false);
        v16h t16;
        #pragma unroll
        for (int i = 0; i < 16; ++i) t16[i] = ap[i] * ap[i];
        v8f e_xx = __builtin_amdgcn_wmma_f32_16x16x32_f16(false, t16, false, bw,
                                                          (short)0, zero, false, false);
        #pragma unroll
        for (int i = 0; i < 16; ++i) t16[i] = aq[i] * aq[i];
        v8f e_yy = __builtin_amdgcn_wmma_f32_16x16x32_f16(false, t16, false, bw,
                                                          (short)0, zero, false, false);
        #pragma unroll
        for (int i = 0; i < 16; ++i) t16[i] = ap[i] * aq[i];
        v8f e_xy = __builtin_amdgcn_wmma_f32_16x16x32_f16(false, t16, false, bw,
                                                          (short)0, zero, false, false);

        // ---- pointwise SSIM on the 8 accumulator elements this lane owns
        #pragma unroll
        for (int i = 0; i < 8; ++i) {
            float mx = mu_x[i], my = mu_y[i];
            float mxx = mx * mx, myy = my * my, mxy = mx * my;
            float sxx = e_xx[i] - mxx;
            float syy = e_yy[i] - myy;
            float sxy = e_xy[i] - mxy;
            float num = (2.f * mxy + SSIM_C1) * (2.f * sxy + SSIM_C2);
            float den = (mxx + myy + SSIM_C1) * (sxx + syy + SSIM_C2);
            acc += num * __builtin_amdgcn_rcpf(den);   // den > 0, ~1 ulp
        }
    }

    // ---- deterministic workgroup tree reduction
    red[tid] = acc;
    __syncthreads();
    #pragma unroll
    for (int s = THREADS / 2; s > 0; s >>= 1) {
        if (tid < s) red[tid] += red[tid + s];
        __syncthreads();
    }
    if (tid == 0) ws[blockIdx.y * gridDim.x + blockIdx.x] = red[0];
}

__global__ __launch_bounds__(256)
void ssim_finalize(const float* __restrict__ ws, float* __restrict__ out, int nparts)
{
    __shared__ float red[256];
    float a = 0.f;
    for (int i = threadIdx.x; i < nparts; i += 256) a += ws[i];  // fixed order
    red[threadIdx.x] = a;
    __syncthreads();
    #pragma unroll
    for (int s = 128; s > 0; s >>= 1) {
        if (threadIdx.x < s) red[threadIdx.x] += red[threadIdx.x + s];
        __syncthreads();
    }
    if (threadIdx.x == 0) out[0] = 1.0f - red[0] * TOTAL_INV;
}

extern "C" void kernel_launch(void* const* d_in, const int* in_sizes, int n_in,
                              void* d_out, int out_size, void* d_ws, size_t ws_size,
                              hipStream_t stream)
{
    const float* pred   = (const float*)d_in[0];
    const float* target = (const float*)d_in[1];
    float* ws  = (float*)d_ws;
    float* out = (float*)d_out;

    dim3 grid(T_LEN / T_CHUNK, N_ROWS / ROWS16);   // 16 x 128 = 2048 workgroups
    ssim_wmma_kernel<<<grid, THREADS, 0, stream>>>(pred, target, ws);

    const int nparts = (T_LEN / T_CHUNK) * (N_ROWS / ROWS16);
    ssim_finalize<<<1, 256, 0, stream>>>(ws, out, nparts);
}
